// Attention_5214090297909
// MI455X (gfx1250) — matrix operand order repaired and checked by reading, not compiled, not run on hardware
//
#include <hip/hip_runtime.h>
#include <hip/hip_bf16.h>

#define BATCH   16
#define LK      2048
#define LQ      2048
#define DIM     256
#define KSLICES 16
#define KSLICE_LEN (LK / KSLICES)   // 128

typedef __attribute__((ext_vector_type(16))) __bf16 v16bf;
typedef __attribute__((ext_vector_type(8)))  __bf16 v8bf;
typedef __attribute__((ext_vector_type(8)))  float  v8f;
typedef __attribute__((ext_vector_type(4)))  float  f32x4;
typedef __attribute__((ext_vector_type(4)))  int    v4i;

union BF16Frag { v16bf v; v8bf h[2]; __bf16 e[16]; };

// ---- feature detection ----
#if defined(__has_builtin)
#if __has_builtin(__builtin_amdgcn_global_load_async_to_lds_b128)
#define HAVE_ASYNC_LDS 1
#endif
#if __has_builtin(__builtin_amdgcn_sched_group_barrier)
#define HAVE_SCHED_GROUP 1
#endif
#endif
#ifndef HAVE_ASYNC_LDS
#define HAVE_ASYNC_LDS 0
#endif
#ifndef HAVE_SCHED_GROUP
#define HAVE_SCHED_GROUP 0
#endif

// sched_group_barrier masks: 0x008 = MFMA/WMMA, 0x020 = VMEM read, 0x100 = DS read
#if HAVE_SCHED_GROUP
#define SCHED_WMMA(n, id)    __builtin_amdgcn_sched_group_barrier(0x008, (n), (id))
#define SCHED_VMEMRD(n, id)  __builtin_amdgcn_sched_group_barrier(0x020, (n), (id))
#define SCHED_DSRD(n, id)    __builtin_amdgcn_sched_group_barrier(0x100, (n), (id))
#else
#define SCHED_WMMA(n, id)
#define SCHED_VMEMRD(n, id)
#define SCHED_DSRD(n, id)
#endif

typedef __attribute__((address_space(1))) v4i as1_v4i;
typedef __attribute__((address_space(3))) v4i as3_v4i;

__device__ __forceinline__ void copy16_to_lds(__bf16* dst_lds, const __bf16* src_global) {
#if HAVE_ASYNC_LDS
  __builtin_amdgcn_global_load_async_to_lds_b128(
      (as1_v4i*)(uintptr_t)src_global,
      (as3_v4i*)(uint32_t)(uintptr_t)dst_lds, 0, 0);
#else
  *(v8bf*)dst_lds = *(const v8bf*)src_global;
#endif
}

__device__ __forceinline__ void wait_async_le2() {
#if HAVE_ASYNC_LDS
  asm volatile("s_wait_asynccnt 0x2" ::: "memory");
#endif
}
__device__ __forceinline__ void wait_async_0() {
#if HAVE_ASYNC_LDS
  asm volatile("s_wait_asynccnt 0x0" ::: "memory");
#endif
}

__device__ __forceinline__ v8f wmma_bf16(v16bf a, v16bf b, v8f c) {
  // (neg_a, A, neg_b, B, c_mod, C, reuse_a, reuse_b)
  return __builtin_amdgcn_wmma_f32_16x16x32_bf16(false, a, false, b, (short)0, c,
                                                 false, false);
}

// A fragment (16x32 bf16, row-major, row stride ldr elements).
// Lanes 0-15: M=lane, K={0..7,16..23}; lanes 16-31: K={8..15,24..31}.
__device__ __forceinline__ v16bf load_a_bf16(const __bf16* __restrict__ base, int ldr, int lane) {
  const __bf16* p = base + (size_t)(lane & 15) * ldr + ((lane < 16) ? 0 : 8);
  BF16Frag u;
  u.h[0] = *(const v8bf*)(p);
  u.h[1] = *(const v8bf*)(p + 16);
  return u.v;
}

// B fragment (32x16 bf16); source row n holds K contiguous (stride ldr).
// Lanes 0-15: N=lane, K={0..7,16..23}; lanes 16-31: N=lane-16, K={8..15,24..31}.
__device__ __forceinline__ v16bf load_b_bf16(const __bf16* __restrict__ base, int ldr, int lane) {
  const __bf16* p = base + (size_t)(lane & 15) * ldr + ((lane < 16) ? 0 : 8);
  BF16Frag u;
  u.h[0] = *(const v8bf*)(p);
  u.h[1] = *(const v8bf*)(p + 16);
  return u.v;
}

#define BPITCH 40   // LDS row pitch (elements): 20 banks/row -> conflict-free, 16B aligned

__device__ __forceinline__ v16bf load_b_lds(const __bf16* base, int lane) {
  const __bf16* p = base + (size_t)(lane & 15) * BPITCH + ((lane < 16) ? 0 : 8);
  BF16Frag u;
  u.h[0] = *(const v8bf*)(p);
  u.h[1] = *(const v8bf*)(p + 16);
  return u.v;
}

// A fragment from f32 row-major source (attn), converted to bf16 in-regs.
__device__ __forceinline__ v16bf load_a_f32(const float* __restrict__ base, int ldr, int lane) {
  const float* p = base + (size_t)(lane & 15) * ldr + ((lane < 16) ? 0 : 8);
  f32x4 x0 = *(const f32x4*)(p);
  f32x4 x1 = *(const f32x4*)(p + 4);
  f32x4 x2 = *(const f32x4*)(p + 16);
  f32x4 x3 = *(const f32x4*)(p + 20);
  BF16Frag u;
#pragma unroll
  for (int i = 0; i < 4; ++i) {
    u.e[i]      = (__bf16)x0[i];
    u.e[4 + i]  = (__bf16)x1[i];
    u.e[8 + i]  = (__bf16)x2[i];
    u.e[12 + i] = (__bf16)x3[i];
  }
  return u.v;
}

// ---------------- conversion / transpose kernels ----------------

__global__ void __launch_bounds__(256) cvt_bf16_kernel(const float* __restrict__ x,
                                                       __bf16* __restrict__ y, int n) {
  int i = blockIdx.x * blockDim.x + threadIdx.x;
  int stride = gridDim.x * blockDim.x;
  for (; i < n; i += stride) y[i] = (__bf16)x[i];
}

// qT[b][d][q] = (bf16) query[b][q][d]; 32x32 LDS tile transpose.
__global__ void __launch_bounds__(256) transpose_cvt_kernel(const float* __restrict__ q,
                                                            __bf16* __restrict__ qT) {
  __shared__ __bf16 tile[32][33];
  const int b  = blockIdx.z;
  const int q0 = blockIdx.x * 32;
  const int d0 = blockIdx.y * 32;
  const int tx = threadIdx.x;   // 0..31
  const int ty = threadIdx.y;   // 0..7
  const float* src = q + ((size_t)b * LQ + q0) * DIM + d0;
#pragma unroll
  for (int j = 0; j < 4; ++j) {
    int r = ty * 4 + j;                               // q_local
    tile[r][tx] = (__bf16)src[(size_t)r * DIM + tx];
  }
  __syncthreads();
  __bf16* dst = qT + ((size_t)b * DIM + d0) * LQ + q0;
#pragma unroll
  for (int j = 0; j < 4; ++j) {
    int r = ty * 4 + j;                               // d_local
    dst[(size_t)r * LQ + tx] = tile[tx][r];
  }
}

// ---------------- kernel 1: S = key @ query^T ----------------
// grid (LQ/128, LK/128, BATCH); wave tile 16(M) x 128(N).
__global__ void __launch_bounds__(256) scores_kernel(const __bf16* __restrict__ kb,
                                                     const __bf16* __restrict__ qb,
                                                     float* __restrict__ attn) {
  const int lane = threadIdx.x & 31;
  const int wave = threadIdx.x >> 5;
  const int b  = blockIdx.z;
  const int m0 = blockIdx.y * 128 + wave * 16;
  const int n0 = blockIdx.x * 128;
  const __bf16* A  = kb + ((size_t)b * LK + m0) * DIM;
  const __bf16* Bq = qb + ((size_t)b * LQ + n0) * DIM;

  v8f acc[8] = {};
#pragma unroll
  for (int d0 = 0; d0 < DIM; d0 += 32) {
    v16bf a = load_a_bf16(A + d0, DIM, lane);
    v16bf bb[8];
#pragma unroll
    for (int nt = 0; nt < 8; ++nt)
      bb[nt] = load_b_bf16(Bq + (size_t)nt * 16 * DIM + d0, DIM, lane);
#pragma unroll
    for (int nt = 0; nt < 8; ++nt)
      acc[nt] = wmma_bf16(a, bb[nt], acc[nt]);
    // steer scheduler: interleave {1 WMMA, ~2.25 VMEM-reads} so the next
    // chunk's 18 loads issue underneath this chunk's 8 WMMAs
#pragma unroll
    for (int g = 0; g < 8; ++g) {
      SCHED_WMMA(1, 0);
      SCHED_VMEMRD(2, 0);
    }
    SCHED_VMEMRD(2, 0);
  }

  float* out = attn + (size_t)b * LK * LQ;
#pragma unroll
  for (int nt = 0; nt < 8; ++nt) {
#pragma unroll
    for (int v = 0; v < 8; ++v) {
      int m = m0 + ((lane < 16) ? v : v + 8);
      int n = n0 + nt * 16 + (lane & 15);
      out[(size_t)m * LQ + n] = acc[nt][v];
    }
  }
}

// ---------------- kernel 2: column softmax over k (split-K online) ----------------

__global__ void __launch_bounds__(256) softmax_partial_kernel(const float* __restrict__ attn,
                                                              float* __restrict__ pm,
                                                              float* __restrict__ ps) {
  const int q = blockIdx.x * 256 + threadIdx.x;
  const int slice = blockIdx.y;
  const int b = blockIdx.z;
  const float* col = attn + (size_t)b * LK * LQ + q;
  const int k0 = slice * KSLICE_LEN;
  float m = -__builtin_inff(), s = 0.f;
#pragma unroll 4
  for (int k = k0; k < k0 + KSLICE_LEN; ++k) {
    float x  = col[(size_t)k * LQ];
    float mn = fmaxf(m, x);
    s = s * __expf(m - mn) + __expf(x - mn);
    m = mn;
  }
  pm[((size_t)b * KSLICES + slice) * LQ + q] = m;
  ps[((size_t)b * KSLICES + slice) * LQ + q] = s;
}

__global__ void __launch_bounds__(256) softmax_combine_kernel(const float* __restrict__ pm,
                                                              const float* __restrict__ ps,
                                                              float* __restrict__ fm,
                                                              float* __restrict__ fs) {
  const int q = blockIdx.x * 256 + threadIdx.x;
  const int b = blockIdx.y;
  float m = -__builtin_inff(), s = 0.f;
#pragma unroll
  for (int sl = 0; sl < KSLICES; ++sl) {
    float mi = pm[((size_t)b * KSLICES + sl) * LQ + q];
    float si = ps[((size_t)b * KSLICES + sl) * LQ + q];
    float mn = fmaxf(m, mi);
    s = s * __expf(m - mn) + si * __expf(mi - mn);
    m = mn;
  }
  fm[(size_t)b * LQ + q] = m;
  fs[(size_t)b * LQ + q] = s;
}

__global__ void __launch_bounds__(256) softmax_norm_kernel(float* __restrict__ attn,
                                                           const float* __restrict__ fm,
                                                           const float* __restrict__ fs) {
  const int q = blockIdx.x * 256 + threadIdx.x;
  const int slice = blockIdx.y;
  const int b = blockIdx.z;
  const float m   = fm[(size_t)b * LQ + q];
  const float inv = 1.f / fs[(size_t)b * LQ + q];
  float* col = attn + (size_t)b * LK * LQ + q;
  const int k0 = slice * KSLICE_LEN;
#pragma unroll 4
  for (int k = k0; k < k0 + KSLICE_LEN; ++k) {
    float x = col[(size_t)k * LQ];
    col[(size_t)k * LQ] = __expf(x - m) * inv;
  }
}

// ---------------- kernel 3: energy = attn @ query ----------------
// grid (DIM/128, LK/128, BATCH). B tile (qT, 128x32) shared by all 8 waves:
// staged in LDS, double-buffered, via async global->LDS copies.
__global__ void __launch_bounds__(256) energy_kernel(const float* __restrict__ attn,
                                                     const __bf16* __restrict__ qT,
                                                     __bf16* __restrict__ energy) {
  __shared__ __bf16 Bs[2][128][BPITCH];
  const int tid  = threadIdx.x;
  const int lane = tid & 31;
  const int wave = tid >> 5;
  const int b  = blockIdx.z;
  const int m0 = blockIdx.y * 128 + wave * 16;   // k-row
  const int n0 = blockIdx.x * 128;               // d-col
  const float*  A  = attn + ((size_t)b * LK + m0) * LQ;
  const __bf16* Bt = qT + ((size_t)b * DIM + n0) * LQ;

  // Stage one 128x32 chunk (8KB): 512 x 16B segments, 2 per thread.
  auto stage = [&](int q0, int buf) {
#pragma unroll
    for (int i = 0; i < 2; ++i) {
      int idx = tid + i * 256;
      int row = idx >> 2;         // 0..127 (d_local)
      int seg = idx & 3;          // 0..3   (16B segment along q)
      copy16_to_lds(&Bs[buf][row][seg * 8],
                    Bt + (size_t)row * LQ + q0 + seg * 8);
    }
  };

  v8f acc[8] = {};
  stage(0, 0);
  constexpr int NIT = LQ / 32;
  for (int it = 0; it < NIT; ++it) {
    const int q0  = it * 32;
    const int buf = it & 1;
    if (it + 1 < NIT) {
      stage(q0 + 32, buf ^ 1);
      wait_async_le2();           // previous chunk's 2 copies complete
    } else {
      wait_async_0();
    }
    __syncthreads();              // all waves' segments visible

    v16bf a = load_a_f32(A + q0, LQ, lane);
    v16bf bb[8];
#pragma unroll
    for (int nt = 0; nt < 8; ++nt)
      bb[nt] = load_b_lds(&Bs[buf][nt * 16][0], lane);
#pragma unroll
    for (int nt = 0; nt < 8; ++nt)
      acc[nt] = wmma_bf16(a, bb[nt], acc[nt]);
    // interleave {1 WMMA, 2 DS-reads}; A-loads (4 VMEM reads) slot in early
    SCHED_VMEMRD(4, 1);
#pragma unroll
    for (int g = 0; g < 8; ++g) {
      SCHED_WMMA(1, 1);
      SCHED_DSRD(2, 1);
    }

    __syncthreads();              // done reading buf before it is restaged
  }

  __bf16* E = energy + (size_t)b * LK * DIM;
#pragma unroll
  for (int nt = 0; nt < 8; ++nt) {
#pragma unroll
    for (int v = 0; v < 8; ++v) {
      int m = m0 + ((lane < 16) ? v : v + 8);
      int n = n0 + nt * 16 + (lane & 15);
      E[(size_t)m * DIM + n] = (__bf16)acc[nt][v];
    }
  }
}

// ---------------- kernel 4: out = tanh([energy|key] @ W^T + b) ----------------
__global__ void __launch_bounds__(256) out_kernel(const __bf16* __restrict__ energy,
                                                  const __bf16* __restrict__ kb,
                                                  const __bf16* __restrict__ Wb,
                                                  const float* __restrict__ bias,
                                                  float* __restrict__ out) {
  const int lane = threadIdx.x & 31;
  const int wave = threadIdx.x >> 5;
  const int m0 = blockIdx.y * 128 + wave * 16;   // flattened [B*LK] row
  const int n0 = blockIdx.x * 128;               // output d
  const __bf16* A1 = energy + (size_t)m0 * DIM;
  const __bf16* A2 = kb + (size_t)m0 * DIM;

  v8f acc[8] = {};
#pragma unroll
  for (int e0 = 0; e0 < DIM; e0 += 32) {         // energy half: W[:, 0..255]
    v16bf a = load_a_bf16(A1 + e0, DIM, lane);
    v16bf bb[8];
#pragma unroll
    for (int nt = 0; nt < 8; ++nt)
      bb[nt] = load_b_bf16(Wb + (size_t)(n0 + nt * 16) * (2 * DIM) + e0, 2 * DIM, lane);
#pragma unroll
    for (int nt = 0; nt < 8; ++nt)
      acc[nt] = wmma_bf16(a, bb[nt], acc[nt]);
#pragma unroll
    for (int g = 0; g < 8; ++g) {
      SCHED_WMMA(1, 2);
      SCHED_VMEMRD(2, 2);
    }
    SCHED_VMEMRD(2, 2);
  }
#pragma unroll
  for (int e0 = 0; e0 < DIM; e0 += 32) {         // key half: W[:, 256..511]
    v16bf a = load_a_bf16(A2 + e0, DIM, lane);
    v16bf bb[8];
#pragma unroll
    for (int nt = 0; nt < 8; ++nt)
      bb[nt] = load_b_bf16(Wb + (size_t)(n0 + nt * 16) * (2 * DIM) + DIM + e0, 2 * DIM, lane);
#pragma unroll
    for (int nt = 0; nt < 8; ++nt)
      acc[nt] = wmma_bf16(a, bb[nt], acc[nt]);
#pragma unroll
    for (int g = 0; g < 8; ++g) {
      SCHED_WMMA(1, 3);
      SCHED_VMEMRD(2, 3);
    }
    SCHED_VMEMRD(2, 3);
  }

#pragma unroll
  for (int nt = 0; nt < 8; ++nt) {
#pragma unroll
    for (int v = 0; v < 8; ++v) {
      int m = m0 + ((lane < 16) ? v : v + 8);
      int n = n0 + nt * 16 + (lane & 15);
      out[(size_t)m * DIM + n] = tanhf(acc[nt][v] + bias[n]);
    }
  }
}

// ---------------- launch ----------------

extern "C" void kernel_launch(void* const* d_in, const int* in_sizes, int n_in,
                              void* d_out, int out_size, void* d_ws, size_t ws_size,
                              hipStream_t stream) {
  (void)in_sizes; (void)n_in; (void)out_size; (void)ws_size;
  const float* key   = (const float*)d_in[0];
  const float* query = (const float*)d_in[1];
  const float* W     = (const float*)d_in[2];
  const float* bias  = (const float*)d_in[3];

  const size_t nKQ = (size_t)BATCH * LK * DIM;   // 8,388,608
  const size_t nW  = (size_t)DIM * 2 * DIM;      // 131,072

  float* out_f    = (float*)d_out;
  float* attn_out = out_f + nKQ;

  char* ws = (char*)d_ws;
  size_t off = 0;
  auto take = [&](size_t bytes) -> void* {
    void* p = ws + off;
    off = (off + bytes + 255) & ~(size_t)255;
    return p;
  };
  __bf16* key_b    = (__bf16*)take(nKQ * 2);
  __bf16* query_b  = (__bf16*)take(nKQ * 2);
  __bf16* qT_b     = (__bf16*)take(nKQ * 2);
  __bf16* W_b      = (__bf16*)take(nW * 2);
  __bf16* energy_b = (__bf16*)take(nKQ * 2);
  float*  pm = (float*)take((size_t)BATCH * KSLICES * LQ * 4);
  float*  ps = (float*)take((size_t)BATCH * KSLICES * LQ * 4);
  float*  fm = (float*)take((size_t)BATCH * LQ * 4);
  float*  fs = (float*)take((size_t)BATCH * LQ * 4);

  cvt_bf16_kernel<<<dim3(1024), dim3(256), 0, stream>>>(key, key_b, (int)nKQ);
  cvt_bf16_kernel<<<dim3(1024), dim3(256), 0, stream>>>(query, query_b, (int)nKQ);
  cvt_bf16_kernel<<<dim3(64),   dim3(256), 0, stream>>>(W, W_b, (int)nW);
  transpose_cvt_kernel<<<dim3(LQ / 32, DIM / 32, BATCH), dim3(32, 8), 0, stream>>>(query, qT_b);

  scores_kernel<<<dim3(LQ / 128, LK / 128, BATCH), dim3(256), 0, stream>>>(key_b, query_b, attn_out);

  softmax_partial_kernel<<<dim3(LQ / 256, KSLICES, BATCH), dim3(256), 0, stream>>>(attn_out, pm, ps);
  softmax_combine_kernel<<<dim3(LQ / 256, BATCH), dim3(256), 0, stream>>>(pm, ps, fm, fs);
  softmax_norm_kernel<<<dim3(LQ / 256, KSLICES, BATCH), dim3(256), 0, stream>>>(attn_out, fm, fs);

  energy_kernel<<<dim3(DIM / 128, LK / 128, BATCH), dim3(256), 0, stream>>>(attn_out, qT_b, energy_b);

  out_kernel<<<dim3(DIM / 128, (BATCH * LK) / 128), dim3(256), 0, stream>>>(energy_b, key_b, W_b, bias, out_f);
}
